// Attention_70557722739259
// MI455X (gfx1250) — matrix operand order repaired and checked by reading, not compiled, not run on hardware
//
#include <hip/hip_runtime.h>
#include <hip/hip_bf16.h>

// ---------------------------------------------------------------------------
// Gated attention for MI455X (gfx1250, wave32, WMMA bf16 16x16x32).
// Pipeline:
//   cvt(x,wq,wk,wv,wo) -> GEMM qg/k/v -> rmsnorm+mrope(q,k)+sigmoid(gate)
//   -> flash attention (LDS tiles, online softmax, WMMA) -> GEMM out proj.
// Data movement: GLOBAL_LOAD_ASYNC_TO_LDS_B128 (ASYNCcnt) when available,
// double-buffered LDS tiles in the GEMM, global_prefetch for the B path.
// ---------------------------------------------------------------------------

typedef float v8f  __attribute__((ext_vector_type(8)));
typedef __bf16 v16bf __attribute__((ext_vector_type(16)));

union Frag  { uint4 u[2]; v16bf v; };
union AccU  { v8f v; float f[8]; };

__device__ __forceinline__ unsigned short f2bf(float f) {
    unsigned u = __float_as_uint(f);
    unsigned r = (u + 0x7FFFu + ((u >> 16) & 1u)) >> 16;   // RNE
    return (unsigned short)r;
}
__device__ __forceinline__ float bf2f(unsigned short h) {
    return __uint_as_float(((unsigned)h) << 16);
}

__device__ __forceinline__ v8f wmma_bf16(const Frag& a, const Frag& b, v8f c) {
    return __builtin_amdgcn_wmma_f32_16x16x32_bf16(
        /*neg_a=*/false, a.v, /*neg_b=*/false, b.v,
        /*c_mod=*/(short)0, c, /*reuse_a=*/false, /*reuse_b=*/false);
}

// ---- CDNA5 async global -> LDS copy (ASYNCcnt), with safe fallback --------
#if defined(__HIP_DEVICE_COMPILE__) && \
    __has_builtin(__builtin_amdgcn_global_load_async_to_lds_b128)
#define USE_ASYNC_COPY 1
#endif

__device__ __forceinline__ void lds_copy_b128(void* lds_dst, const void* gsrc) {
#ifdef USE_ASYNC_COPY
    typedef int v4i_ __attribute__((vector_size(16)));
    __builtin_amdgcn_global_load_async_to_lds_b128(
        (__attribute__((address_space(1))) v4i_*)(v4i_*)(void*)gsrc,
        (__attribute__((address_space(3))) v4i_*)(v4i_*)lds_dst,
        /*imm offset=*/0, /*cpol=*/0);
#else
    *(uint4*)lds_dst = *(const uint4*)gsrc;
#endif
}

__device__ __forceinline__ void wait_async_lds() {
#ifdef USE_ASYNC_COPY
    asm volatile("s_wait_asynccnt 0" ::: "memory");
#endif
}

// Problem constants
#define CB   2
#define CT   2048
#define CD   2048
#define CNH  16
#define CKH  2
#define CH   256
#define CBT  4096            // B*T
#define K_MASK_F (-2.3819763e+38f)

// ---------------------------------------------------------------------------
// f32 -> bf16 convert (grid-stride)
// ---------------------------------------------------------------------------
__global__ void cvt_bf16_kernel(const float* __restrict__ in,
                                unsigned short* __restrict__ out,
                                long long n) {
    long long i = (long long)blockIdx.x * blockDim.x + threadIdx.x;
    long long stride = (long long)gridDim.x * blockDim.x;
    for (; i < n; i += stride) out[i] = f2bf(in[i]);
}

// ---------------------------------------------------------------------------
// Generic bf16 WMMA GEMM: C[MxN] = A[MxK] * B[KxN], fp32 accumulate.
// Block tile 128x128x32, 256 threads = 8 waves (2x4), wave tile 64x32.
// Double-buffered LDS tiles; A staged via async-to-LDS, B transposed via
// VGPR round-trip (needed for the B-operand layout) with prefetch.
// mode 0: store f32 row-major.  mode 1: store bf16 transposed V layout
//         (row=(b,t), col=(kh,dim)) -> out[((b*KH+kh)*H+dim)*T + t].
// ---------------------------------------------------------------------------
__global__ __launch_bounds__(256) void gemm_bf16_wmma(
    const unsigned short* __restrict__ A,
    const unsigned short* __restrict__ Bm,
    void* __restrict__ Cout,
    int M, int Nn, int Kd, int mode)
{
    __shared__ __align__(16) unsigned short sA[2][128][32];   // row-major A
    __shared__ __align__(16) unsigned short sB[2][128][32];   // sB[n][k]

    const int tid  = threadIdx.x;
    const int lane = tid & 31;
    const int w    = tid >> 5;
    const int wm   = w >> 2;            // 0..1
    const int wn   = w & 3;             // 0..3
    const int half = lane >> 4;
    const int l16  = lane & 15;
    const int m0   = blockIdx.y * 128;
    const int n0   = blockIdx.x * 128;

    const int arow = tid >> 1;          // A-stage mapping
    const int aks  = (tid & 1) * 16;
    const int bcol = tid & 127;         // B-stage mapping
    const int bkb  = (tid >> 7) * 16;

    v8f acc[4][2];
    #pragma unroll
    for (int i = 0; i < 4; i++)
        #pragma unroll
        for (int j = 0; j < 2; j++) acc[i][j] = 0.0f;

    auto stage = [&](int buf, int k0) {
        const unsigned short* ga = A + (size_t)(m0 + arow) * Kd + k0 + aks;
        lds_copy_b128(&sA[buf][arow][aks],     ga);
        lds_copy_b128(&sA[buf][arow][aks + 8], ga + 8);
        unsigned short tmp[16];
        #pragma unroll
        for (int j = 0; j < 16; j++)
            tmp[j] = Bm[(size_t)(k0 + bkb + j) * Nn + n0 + bcol];
        *(uint4*)&sB[buf][bcol][bkb]     = *(uint4*)&tmp[0];
        *(uint4*)&sB[buf][bcol][bkb + 8] = *(uint4*)&tmp[8];
    };

    stage(0, 0);
    wait_async_lds();
    __syncthreads();

    const int nk = Kd >> 5;
    for (int ki = 0; ki < nk; ki++) {
        const int cur = ki & 1;
        if (ki + 1 < nk) stage(cur ^ 1, (ki + 1) << 5);
        if (ki + 2 < nk) {   // prefetch B rows two tiles ahead (cacheline pull)
            __builtin_prefetch(
                Bm + (size_t)(((ki + 2) << 5) + bkb) * Nn + n0 + bcol, 0, 1);
        }

        Frag bfr[2];
        #pragma unroll
        for (int nt = 0; nt < 2; nt++) {
            int col = wn * 32 + nt * 16 + l16;
            bfr[nt].u[0] = *(const uint4*)&sB[cur][col][half * 16];
            bfr[nt].u[1] = *(const uint4*)&sB[cur][col][half * 16 + 8];
        }
        #pragma unroll
        for (int mt = 0; mt < 4; mt++) {
            Frag af;
            int row = wm * 64 + mt * 16 + l16;
            int klo = half * 16;
            af.u[0] = *(const uint4*)&sA[cur][row][klo];
            af.u[1] = *(const uint4*)&sA[cur][row][klo + 8];
            #pragma unroll
            for (int nt = 0; nt < 2; nt++)
                acc[mt][nt] = wmma_bf16(af, bfr[nt], acc[mt][nt]);
        }

        wait_async_lds();
        __syncthreads();
    }

    #pragma unroll
    for (int mt = 0; mt < 4; mt++) {
        #pragma unroll
        for (int nt = 0; nt < 2; nt++) {
            AccU a; a.v = acc[mt][nt];
            int col = n0 + wn * 32 + nt * 16 + l16;
            #pragma unroll
            for (int r = 0; r < 8; r++) {
                int row = m0 + wm * 64 + mt * 16 + half * 8 + r;
                if (mode == 0) {
                    ((float*)Cout)[(size_t)row * Nn + col] = a.f[r];
                } else {
                    int b  = row >> 11, t  = row & 2047;      // T = 2048
                    int kh = col >> 8,  dm = col & 255;       // H = 256
                    ((unsigned short*)Cout)
                        [(((size_t)(b * CKH + kh) * CH + dm) << 11) + t] =
                        f2bf(a.f[r]);
                }
            }
        }
    }
}

// ---------------------------------------------------------------------------
// RMS-norm + partial mRoPE for Q (+ sigmoid gate).  One wave per (b,t,n) row.
// Lane holds dims {lane + 32*j}; rotary pair (lane, lane+32) lives in-lane.
// axis = lane % 3 reproduces h_idx/w_idx for MROPE_SECTION = (11,11,10).
// ---------------------------------------------------------------------------
__global__ __launch_bounds__(128) void norm_rope_q_kernel(
    const float* __restrict__ qg,            // (B*T, N*512)
    const int*   __restrict__ positions,     // (3, B, T)
    const float* __restrict__ qw,            // (256,)
    unsigned short* __restrict__ qbuf,       // (B,N,T,H) bf16
    unsigned short* __restrict__ gatebuf)    // (B,N,T,H) bf16 sigmoid(gate)
{
    int row  = blockIdx.x * 4 + (threadIdx.x >> 5);   // 0..B*T*N-1
    int lane = threadIdx.x & 31;
    int bt = row >> 4, n = row & 15;
    const float* src = qg + (size_t)bt * (CNH * 2 * CH) + n * (2 * CH);

    float v[8], ss = 0.f;
    #pragma unroll
    for (int j = 0; j < 8; j++) { v[j] = src[lane + 32 * j]; ss += v[j] * v[j]; }
    ss += __shfl_xor(ss, 1);  ss += __shfl_xor(ss, 2);
    ss += __shfl_xor(ss, 4);  ss += __shfl_xor(ss, 8);  ss += __shfl_xor(ss, 16);
    float rinv = rsqrtf(ss * (1.0f / 256.0f) + 1e-6f);
    #pragma unroll
    for (int j = 0; j < 8; j++)
        v[j] = v[j] * rinv * (1.0f + qw[lane + 32 * j]);

    int axis  = lane % 3;
    float pos = (float)positions[axis * (CB * CT) + bt];
    float ts  = __powf(1.0e6f, (float)lane * (1.0f / 32.0f));
    float fr  = pos / ts;
    float sn = __sinf(fr), cs = __cosf(fr);
    float r0 = v[0] * cs - v[1] * sn;
    float r1 = v[1] * cs + v[0] * sn;
    v[0] = r0; v[1] = r1;

    int b = bt >> 11, t = bt & 2047;
    unsigned short* dst = qbuf    + ((size_t)(b * CNH + n) * CT + t) * CH;
    unsigned short* gd  = gatebuf + ((size_t)(b * CNH + n) * CT + t) * CH;
    #pragma unroll
    for (int j = 0; j < 8; j++) dst[lane + 32 * j] = f2bf(v[j]);
    #pragma unroll
    for (int j = 0; j < 8; j++) {
        float g = src[CH + lane + 32 * j];
        gd[lane + 32 * j] = f2bf(1.0f / (1.0f + __expf(-g)));
    }
}

// Same for K (no gate).  Rows over (b,t,kh).
__global__ __launch_bounds__(128) void norm_rope_k_kernel(
    const float* __restrict__ ktmp,          // (B*T, KH*H)
    const int*   __restrict__ positions,
    const float* __restrict__ kw,
    unsigned short* __restrict__ kbuf)       // (B,KH,T,H) bf16
{
    int row  = blockIdx.x * 4 + (threadIdx.x >> 5);   // 0..B*T*KH-1
    int lane = threadIdx.x & 31;
    int bt = row >> 1, kh = row & 1;
    const float* src = ktmp + (size_t)bt * (CKH * CH) + kh * CH;

    float v[8], ss = 0.f;
    #pragma unroll
    for (int j = 0; j < 8; j++) { v[j] = src[lane + 32 * j]; ss += v[j] * v[j]; }
    ss += __shfl_xor(ss, 1);  ss += __shfl_xor(ss, 2);
    ss += __shfl_xor(ss, 4);  ss += __shfl_xor(ss, 8);  ss += __shfl_xor(ss, 16);
    float rinv = rsqrtf(ss * (1.0f / 256.0f) + 1e-6f);
    #pragma unroll
    for (int j = 0; j < 8; j++)
        v[j] = v[j] * rinv * (1.0f + kw[lane + 32 * j]);

    int axis  = lane % 3;
    float pos = (float)positions[axis * (CB * CT) + bt];
    float ts  = __powf(1.0e6f, (float)lane * (1.0f / 32.0f));
    float fr  = pos / ts;
    float sn = __sinf(fr), cs = __cosf(fr);
    float r0 = v[0] * cs - v[1] * sn;
    float r1 = v[1] * cs + v[0] * sn;
    v[0] = r0; v[1] = r1;

    int b = bt >> 11, t = bt & 2047;
    unsigned short* dst = kbuf + ((size_t)(b * CKH + kh) * CT + t) * CH;
    #pragma unroll
    for (int j = 0; j < 8; j++) dst[lane + 32 * j] = f2bf(v[j]);
}

// ---------------------------------------------------------------------------
// Flash attention.  grid = (B*N, T/64), block = 128 (4 waves x 16 q-rows).
// K tile 64x256 row-major, V tile pre-transposed 256x64 -> async-staged LDS.
// S = Q K^T via WMMA, online softmax (wave32 xor shuffles within 16-lane
// halves, matching the C-layout), P via per-wave LDS relayout, O += P V.
// ---------------------------------------------------------------------------
__global__ __launch_bounds__(128) void flash_attn_kernel(
    const unsigned short* __restrict__ qbuf,     // (B,N,T,H)
    const unsigned short* __restrict__ kbuf,     // (B,KH,T,H)
    const unsigned short* __restrict__ vbuf,     // (B,KH,H,T)
    const unsigned short* __restrict__ gatebuf,  // (B,N,T,H) sigmoid(gate)
    unsigned short* __restrict__ attn_out)       // (B*T, N*H)
{
    __shared__ __align__(16) unsigned short Kt[64][256];
    __shared__ __align__(16) unsigned short Vt[256][64];
    __shared__ __align__(16) unsigned short Pt[4][16][64];

    const int bx = blockIdx.x;               // b*N + n
    const int b  = bx >> 4, n = bx & 15;
    const int kh = n >> 3;                   // 8 q-heads per kv-head
    const int qb = blockIdx.y;               // q block of 64 rows
    const int tid  = threadIdx.x;
    const int lane = tid & 31;
    const int w    = tid >> 5;
    const int half = lane >> 4;
    const int l16  = lane & 15;

    const unsigned short* qrow =
        qbuf + ((size_t)(b * CNH + n) * CT + qb * 64 + w * 16 + l16) * CH;

    v8f O[16];
    #pragma unroll
    for (int i = 0; i < 16; i++) O[i] = 0.0f;
    float mrow[8], lrow[8];
    #pragma unroll
    for (int r = 0; r < 8; r++) { mrow[r] = -3.0e38f; lrow[r] = 0.0f; }

    const int tq_base = qb * 64 + w * 16 + half * 8;
    const int nblocks = qb + 1;              // causal

    for (int sb = 0; sb < nblocks; sb++) {
        __syncthreads();                     // previous-tile reads complete
        {   // K tile: 2 threads per key row, 128 dims (16x async b128) each
            int i = tid >> 1, dbase = (tid & 1) * 128;
            const unsigned short* src =
                kbuf + ((size_t)(b * CKH + kh) * CT + sb * 64 + i) * CH + dbase;
            #pragma unroll
            for (int j = 0; j < 16; j++)
                lds_copy_b128(&Kt[i][dbase + 8 * j], src + 8 * j);
        }
        {   // V tile (pre-transposed): 2 dim rows per thread, 8x async b128
            #pragma unroll
            for (int rr = 0; rr < 2; rr++) {
                int d = rr * 128 + tid;
                const unsigned short* src =
                    vbuf + ((size_t)(b * CKH + kh) * CH + d) * CT + sb * 64;
                #pragma unroll
                for (int j = 0; j < 8; j++)
                    lds_copy_b128(&Vt[d][8 * j], src + 8 * j);
            }
        }
        wait_async_lds();
        __syncthreads();

        // ---- S = Q K^T (16x64 per wave) ----
        v8f sacc[4];
        #pragma unroll
        for (int nt = 0; nt < 4; nt++) sacc[nt] = 0.0f;
        #pragma unroll
        for (int kc = 0; kc < 8; kc++) {
            Frag qf;
            int klo = half * 16;
            qf.u[0] = *(const uint4*)(qrow + kc * 32 + klo);
            qf.u[1] = *(const uint4*)(qrow + kc * 32 + klo + 8);
            #pragma unroll
            for (int nt = 0; nt < 4; nt++) {
                Frag kf;
                int key = nt * 16 + l16;
                kf.u[0] = *(const uint4*)&Kt[key][kc * 32 + half * 16];
                kf.u[1] = *(const uint4*)&Kt[key][kc * 32 + half * 16 + 8];
                sacc[nt] = wmma_bf16(qf, kf, sacc[nt]);
            }
        }

        // ---- scale + causal mask + online softmax ----
        float pv[4][8];
        #pragma unroll
        for (int nt = 0; nt < 4; nt++) {
            AccU a; a.v = sacc[nt];
            int sc = sb * 64 + nt * 16 + l16;
            #pragma unroll
            for (int r = 0; r < 8; r++) {
                float s = a.f[r] * 0.0625f;                  // 256^-0.5
                pv[nt][r] = (sc <= tq_base + r) ? s : K_MASK_F;
            }
        }
        #pragma unroll
        for (int r = 0; r < 8; r++) {
            float mx = fmaxf(fmaxf(pv[0][r], pv[1][r]),
                             fmaxf(pv[2][r], pv[3][r]));
            mx = fmaxf(mx, __shfl_xor(mx, 1));
            mx = fmaxf(mx, __shfl_xor(mx, 2));
            mx = fmaxf(mx, __shfl_xor(mx, 4));
            mx = fmaxf(mx, __shfl_xor(mx, 8));
            float mnew  = fmaxf(mrow[r], mx);
            float alpha = __expf(mrow[r] - mnew);
            float rs = 0.f;
            #pragma unroll
            for (int nt = 0; nt < 4; nt++) {
                float p = __expf(pv[nt][r] - mnew);
                pv[nt][r] = p;
                rs += p;
            }
            rs += __shfl_xor(rs, 1); rs += __shfl_xor(rs, 2);
            rs += __shfl_xor(rs, 4); rs += __shfl_xor(rs, 8);
            lrow[r] = lrow[r] * alpha + rs;
            mrow[r] = mnew;
            #pragma unroll
            for (int dt = 0; dt < 16; dt++) O[dt][r] *= alpha;
        }

        // ---- relayout P through per-wave LDS (C-layout -> A-layout) ----
        #pragma unroll
        for (int nt = 0; nt < 4; nt++)
            #pragma unroll
            for (int r = 0; r < 8; r++)
                Pt[w][half * 8 + r][nt * 16 + l16] = f2bf(pv[nt][r]);

        // ---- O += P V ----
        #pragma unroll
        for (int kb2 = 0; kb2 < 2; kb2++) {
            Frag pf;
            int klo = half * 16;
            pf.u[0] = *(const uint4*)&Pt[w][l16][kb2 * 32 + klo];
            pf.u[1] = *(const uint4*)&Pt[w][l16][kb2 * 32 + klo + 8];
            #pragma unroll
            for (int dt = 0; dt < 16; dt++) {
                Frag vf;
                int dim = dt * 16 + l16;
                vf.u[0] = *(const uint4*)&Vt[dim][kb2 * 32 + half * 16];
                vf.u[1] = *(const uint4*)&Vt[dim][kb2 * 32 + half * 16 + 8];
                O[dt] = wmma_bf16(pf, vf, O[dt]);
            }
        }
    }

    // ---- epilogue: normalize, gate, store ----
    float invl[8];
    #pragma unroll
    for (int r = 0; r < 8; r++) invl[r] = 1.0f / lrow[r];
    const int tq0 = qb * 64 + w * 16 + half * 8;
    #pragma unroll
    for (int dt = 0; dt < 16; dt++) {
        AccU a; a.v = O[dt];
        int dim = dt * 16 + l16;
        #pragma unroll
        for (int r = 0; r < 8; r++) {
            int t = tq0 + r;
            float g = bf2f(gatebuf[((size_t)(b * CNH + n) * CT + t) * CH + dim]);
            float o = a.f[r] * invl[r] * g;
            attn_out[((size_t)(b * CT + t)) * (CNH * CH) + n * CH + dim] = f2bf(o);
        }
    }
}

// ---------------------------------------------------------------------------
// Host side
// ---------------------------------------------------------------------------
extern "C" void kernel_launch(void* const* d_in, const int* in_sizes, int n_in,
                              void* d_out, int out_size, void* d_ws, size_t ws_size,
                              hipStream_t stream) {
    (void)in_sizes; (void)n_in; (void)out_size; (void)ws_size;
    const float* x  = (const float*)d_in[0];   // (B,T,D)
    const int*   pos= (const int*)  d_in[1];   // (3,B,T)
    /* d_in[2] attn_mask: causal tril, not needed */
    const float* wq = (const float*)d_in[3];   // (D, N*2H) = (2048, 8192)
    const float* wk = (const float*)d_in[4];   // (2048, 512)
    const float* wv = (const float*)d_in[5];   // (2048, 512)
    const float* wo = (const float*)d_in[6];   // (4096, 2048)
    const float* qw = (const float*)d_in[7];   // (256,)
    const float* kw = (const float*)d_in[8];   // (256,)
    float* out = (float*)d_out;                // (B,T,D) f32

    // workspace carve (256B aligned slices)
    size_t off = 0;
    auto carve = [&](size_t bytes) -> char* {
        char* p = (char*)d_ws + off;
        off += (bytes + 255) & ~(size_t)255;
        return p;
    };
    unsigned short* x_bf   = (unsigned short*)carve((size_t)CBT * CD * 2);
    unsigned short* wq_bf  = (unsigned short*)carve((size_t)CD * 8192 * 2);
    unsigned short* wk_bf  = (unsigned short*)carve((size_t)CD * 512 * 2);
    unsigned short* wv_bf  = (unsigned short*)carve((size_t)CD * 512 * 2);
    unsigned short* wo_bf  = (unsigned short*)carve((size_t)4096 * CD * 2);
    float*          qg     = (float*)         carve((size_t)CBT * 8192 * 4);
    float*          ktmp   = (float*)         carve((size_t)CBT * 512 * 4);
    unsigned short* qbuf   = (unsigned short*)carve((size_t)CB * CNH * CT * CH * 2);
    unsigned short* gate   = (unsigned short*)carve((size_t)CB * CNH * CT * CH * 2);
    unsigned short* kbuf   = (unsigned short*)carve((size_t)CB * CKH * CT * CH * 2);
    unsigned short* vbuf   = (unsigned short*)carve((size_t)CB * CKH * CT * CH * 2);
    unsigned short* attn   = (unsigned short*)carve((size_t)CBT * CNH * CH * 2);

    // 1) f32 -> bf16 conversions
    cvt_bf16_kernel<<<2048, 256, 0, stream>>>(x,  x_bf,  (long long)CBT * CD);
    cvt_bf16_kernel<<<2048, 256, 0, stream>>>(wq, wq_bf, (long long)CD * 8192);
    cvt_bf16_kernel<<<512,  256, 0, stream>>>(wk, wk_bf, (long long)CD * 512);
    cvt_bf16_kernel<<<512,  256, 0, stream>>>(wv, wv_bf, (long long)CD * 512);
    cvt_bf16_kernel<<<2048, 256, 0, stream>>>(wo, wo_bf, (long long)4096 * CD);

    // 2) projections
    gemm_bf16_wmma<<<dim3(8192 / 128, CBT / 128), 256, 0, stream>>>(
        x_bf, wq_bf, qg,   CBT, 8192, CD, 0);
    gemm_bf16_wmma<<<dim3(512 / 128, CBT / 128), 256, 0, stream>>>(
        x_bf, wk_bf, ktmp, CBT, 512,  CD, 0);
    gemm_bf16_wmma<<<dim3(512 / 128, CBT / 128), 256, 0, stream>>>(
        x_bf, wv_bf, vbuf, CBT, 512,  CD, 1);   // transposed bf16 V

    // 3) norm + rope (+ gate sigmoid)
    norm_rope_q_kernel<<<(CB * CT * CNH) / 4, 128, 0, stream>>>(
        qg, pos, qw, qbuf, gate);
    norm_rope_k_kernel<<<(CB * CT * CKH) / 4, 128, 0, stream>>>(
        ktmp, pos, kw, kbuf);

    // 4) flash attention
    flash_attn_kernel<<<dim3(CB * CNH, CT / 64), 128, 0, stream>>>(
        qbuf, kbuf, vbuf, gate, attn);

    // 5) output projection -> f32 out
    gemm_bf16_wmma<<<dim3(CD / 128, CBT / 128), 256, 0, stream>>>(
        attn, wo_bf, out, CBT, CD, CNH * CH, 0);
}
